// Causalattention_541165879392
// MI455X (gfx1250) — matrix operand order repaired and checked by reading, not compiled, not run on hardware
//
#include <hip/hip_runtime.h>
#include <hip/hip_bf16.h>
#include <math.h>

// Problem constants (from reference): B=4, T=4096, D_IN=D_OUT=768
#define BATCH 4
#define TDIM  4096
#define DDIM  768

typedef __attribute__((ext_vector_type(16))) _Float16 v16h;
typedef __attribute__((ext_vector_type(8)))  _Float16 h8;
typedef __attribute__((ext_vector_type(8)))  float    v8f;
typedef __attribute__((ext_vector_type(4)))  float    f4;

// ---- helpers -------------------------------------------------------------

__device__ __forceinline__ v16h cat16(h8 lo, h8 hi) {
    return __builtin_shufflevector(lo, hi, 0,1,2,3,4,5,6,7,8,9,10,11,12,13,14,15);
}

// load 8 contiguous f32 and convert to 8 f16 (two b128 loads + pk cvt)
__device__ __forceinline__ h8 cvt8(const float* p) {
    f4 a = *(const f4*)p;
    f4 b = *(const f4*)(p + 4);
    h8 r;
    r[0] = (_Float16)a[0]; r[1] = (_Float16)a[1];
    r[2] = (_Float16)a[2]; r[3] = (_Float16)a[3];
    r[4] = (_Float16)b[0]; r[5] = (_Float16)b[1];
    r[6] = (_Float16)b[2]; r[7] = (_Float16)b[3];
    return r;
}

// B-matrix fragment: 16 contiguous f16 (lane n holds K-run at +16*hi)
__device__ __forceinline__ v16h ld_bfrag(const _Float16* p) {
    h8 lo = *(const h8*)p;
    h8 hi = *(const h8*)(p + 8);
    return cat16(lo, hi);
}

// A-matrix fragment: 16 contiguous f16 (lane m holds K-run at +16*hi)
__device__ __forceinline__ v16h ld_afrag(const _Float16* p) {
    h8 lo = *(const h8*)p;
    h8 hi = *(const h8*)(p + 8);
    return cat16(lo, hi);
}

#define WMMA_F16(a, b, c) \
    __builtin_amdgcn_wmma_f32_16x16x32_f16(false, (a), false, (b), (short)0, (c), false, false)

// ---- kernel 0: one-pass fp32 -> f16 conversion (memory-bound) ------
__global__ __launch_bounds__(256, 1)
void cvt_f16_kernel(const float* __restrict__ src, _Float16* __restrict__ dst,
                    int n8) {
    const int i = blockIdx.x * 256 + threadIdx.x;
    if (i < n8) {
        h8 v = cvt8(src + (size_t)i * 8);
        *(h8*)(dst + (size_t)i * 8) = v;
    }
}

// ---- kernel 1: fused QKV projection (pure f16 loads + WMMA) --------------
// grid: (6, 1024, 3) block: 256 (8 waves); wave computes one 16x16 tile.
// Inner loop: 4 b128 loads per v_wmma, two independent accumulator chains
// to hide the f16-WMMA RAW hazard (1 NOP + 4 coexec).
__global__ __launch_bounds__(256, 1)
void qkv_proj_kernel(const _Float16* __restrict__ xh,
                     const _Float16* __restrict__ wh,   // [3][768][768] f16
                     _Float16* __restrict__ qh,
                     _Float16* __restrict__ kh,
                     _Float16* __restrict__ vT) {
    const int lane = threadIdx.x & 31;
    const int wid  = threadIdx.x >> 5;
    const int ct   = blockIdx.x * 8 + wid;   // 0..47 column tile of d_out
    const int rt   = blockIdx.y;             // 0..1023 row tile of B*T
    const int which = blockIdx.z;            // 0:q 1:k 2:v

    const int hi = lane >> 4;
    const int lx = lane & 15;
    const int row0 = rt * 16;

    // A: lane holds x-row m=lx; K-run at d0+16*hi
    const _Float16* xrow = xh + (size_t)(row0 + lx) * DDIM;
    // B: lane holds W row (output col) n = ct*16+lx; K-run at d0+16*hi
    const _Float16* wrow = wh + (size_t)which * DDIM * DDIM
                              + (size_t)(ct * 16 + lx) * DDIM;
    const int abase = hi * 16;
    const int bbase = hi * 16;

    v8f acc0 = {}, acc1 = {};
    #pragma unroll 4
    for (int kk = 0; kk < DDIM; kk += 64) {
        v16h a0 = ld_afrag(xrow + kk + abase);
        v16h b0 = ld_bfrag(wrow + kk + bbase);
        v16h a1 = ld_afrag(xrow + kk + 32 + abase);
        v16h b1 = ld_bfrag(wrow + kk + 32 + bbase);
        acc0 = WMMA_F16(a0, b0, acc0);
        acc1 = WMMA_F16(a1, b1, acc1);
    }
    v8f acc;
    #pragma unroll
    for (int r = 0; r < 8; r++) acc[r] = acc0[r] + acc1[r];

    if (which == 2) {
        // transposed store: vT[b][o][t]; lane's 8 rows are contiguous t
        const int b = row0 / TDIM;
        const int tloc = row0 - b * TDIM;
        _Float16* dst = vT + ((size_t)b * DDIM + (size_t)(ct * 16 + lx)) * TDIM
                           + tloc + 8 * hi;
        h8 vals;
        #pragma unroll
        for (int r = 0; r < 8; r++) vals[r] = (_Float16)acc[r];
        *(h8*)dst = vals;
    } else {
        // fold log2(e)/sqrt(D) into q so attention can use exp2 directly
        const float cs = (which == 0) ? (1.4426950408889634f / 27.712812921102035f)
                                      : 1.0f;
        _Float16* dst = (which == 0) ? qh : kh;
        #pragma unroll
        for (int r = 0; r < 8; r++) {
            const int t = row0 + r + 8 * hi;  // C layout: row = r + 8*hi
            dst[(size_t)t * DDIM + ct * 16 + lx] = (_Float16)(acc[r] * cs);
        }
    }
}

// ---- kernel 2: causal flash attention (online softmax, all-WMMA) ---------
// grid: (T/16, B) block: 32 (one wave). Wave owns 16 query rows x full 768
// output width: 48 f32 accumulator tiles (384 VGPRs) + 24 resident Q
// fragments (192 VGPRs) at 1 wave/SIMD occupancy.
__global__ __launch_bounds__(32, 1)
void flash_attn_kernel(const _Float16* __restrict__ qh,
                       const _Float16* __restrict__ kh,
                       const _Float16* __restrict__ vT,
                       float* __restrict__ out) {
    __shared__ alignas(32) _Float16 pl[16 * 32];  // P tile staging C->A layout

    const int lane = threadIdx.x & 31;
    const int hi = lane >> 4;
    const int lx = lane & 15;
    const int qt = blockIdx.x;   // query tile 0..255
    const int b  = blockIdx.y;

    const _Float16* qb = qh + (size_t)b * TDIM * DDIM;
    const _Float16* kb = kh + (size_t)b * TDIM * DDIM;
    const _Float16* vb = vT + (size_t)b * DDIM * TDIM;
    float* ob = out + (size_t)b * TDIM * DDIM;

    v8f oacc[48];
    #pragma unroll
    for (int i = 0; i < 48; i++) oacc[i] = (v8f){};
    float mrow[8], lrow[8];
    #pragma unroll
    for (int r = 0; r < 8; r++) { mrow[r] = -__builtin_inff(); lrow[r] = 0.0f; }

    const int qrow0 = qt * 16;
    const int abase = hi * 16;

    // Hoist Q: 24 A-fragments resident for the whole key loop.
    const _Float16* qrow = qb + (size_t)(qrow0 + lx) * DDIM;  // A: row m = lx
    v16h qf[24];
    #pragma unroll
    for (int kk = 0; kk < 24; kk++) qf[kk] = ld_afrag(qrow + kk * 32 + abase);

    const int nsteps = qt / 2 + 1;  // 32 keys per step, causal bound
    for (int s = 0; s < nsteps; s++) {
        const int j0 = s * 32;

        // ---- S = q . k^T for two 16-key tiles (two interleaved chains) --
        v8f s0 = {}, s1 = {};
        const _Float16* kr0 = kb + (size_t)(j0 + lx) * DDIM + hi * 16;
        const _Float16* kr1 = kb + (size_t)(j0 + 16 + lx) * DDIM + hi * 16;
        #pragma unroll 4
        for (int kk = 0; kk < 24; kk++) {
            v16h b0 = ld_bfrag(kr0 + kk * 32);
            v16h b1 = ld_bfrag(kr1 + kk * 32);
            s0 = WMMA_F16(qf[kk], b0, s0);
            s1 = WMMA_F16(qf[kk], b1, s1);
        }

        // ---- causal mask + online softmax (base-2; scale folded into q) -
        float alpha[8];
        const int n0 = j0 + lx;
        const int n1 = n0 + 16;
        #pragma unroll
        for (int r = 0; r < 8; r++) {
            const int m = qrow0 + r + 8 * hi;  // global query row this elem
            float e0 = (n0 <= m) ? s0[r] : -__builtin_inff();
            float e1 = (n1 <= m) ? s1[r] : -__builtin_inff();
            // row max across the 16-lane column group
            float t = fmaxf(e0, e1);
            t = fmaxf(t, __shfl_xor(t, 1, 32));
            t = fmaxf(t, __shfl_xor(t, 2, 32));
            t = fmaxf(t, __shfl_xor(t, 4, 32));
            t = fmaxf(t, __shfl_xor(t, 8, 32));
            const float nm = fmaxf(mrow[r], t);
            const float al = exp2f(mrow[r] - nm);
            mrow[r] = nm;
            alpha[r] = al;
            const float p0 = exp2f(e0 - nm);
            const float p1 = exp2f(e1 - nm);
            float ps = p0 + p1;
            ps += __shfl_xor(ps, 1, 32);
            ps += __shfl_xor(ps, 2, 32);
            ps += __shfl_xor(ps, 4, 32);
            ps += __shfl_xor(ps, 8, 32);
            lrow[r] = lrow[r] * al + ps;
            // stage P into LDS as [m_local][j_local] for A-layout reload
            const int ml = r + 8 * hi;
            pl[ml * 32 + lx]      = (_Float16)p0;
            pl[ml * 32 + 16 + lx] = (_Float16)p1;
        }
        asm volatile("s_wait_dscnt 0" ::: "memory");
        __builtin_amdgcn_wave_barrier();

        // reload P in A-matrix layout: row lx, K-run at abase (contig)
        const v16h pA = ld_afrag(&pl[lx * 32 + abase]);
        __builtin_amdgcn_wave_barrier();

        // ---- rescale O and accumulate P.V (48 independent chains) ------
        #pragma unroll
        for (int ctc = 0; ctc < 48; ctc++) {
            v8f o = oacc[ctc];
            #pragma unroll
            for (int r = 0; r < 8; r++) o[r] *= alpha[r];
            // B frag from vT: lane col n = ctc*16+lx, K-run j0+16*hi (contig)
            const _Float16* vrow = vb + (size_t)(ctc * 16 + lx) * TDIM + j0 + hi * 16;
            v16h bv = ld_bfrag(vrow);
            oacc[ctc] = WMMA_F16(pA, bv, o);
        }
    }

    // ---- normalize and store (fp32 out) ---------------------------------
    float inv[8];
    #pragma unroll
    for (int r = 0; r < 8; r++) inv[r] = 1.0f / lrow[r];
    #pragma unroll
    for (int ctc = 0; ctc < 48; ctc++) {
        #pragma unroll
        for (int r = 0; r < 8; r++) {
            ob[(size_t)(qrow0 + r + 8 * hi) * DDIM + ctc * 16 + lx]
                = oacc[ctc][r] * inv[r];
        }
    }
}

// ---- launcher ------------------------------------------------------------
extern "C" void kernel_launch(void* const* d_in, const int* in_sizes, int n_in,
                              void* d_out, int out_size, void* d_ws, size_t ws_size,
                              hipStream_t stream) {
    const float* x  = (const float*)d_in[0];
    const float* Wq = (const float*)d_in[1];
    const float* Wk = (const float*)d_in[2];
    const float* Wv = (const float*)d_in[3];
    float* out = (float*)d_out;

    // workspace layout (f16): xh | Wh[3] | q | k | vT  (~100 MiB total)
    const size_t nX = (size_t)BATCH * TDIM * DDIM;   // 12,582,912
    const size_t nW = (size_t)DDIM * DDIM;           //    589,824
    _Float16* xh = (_Float16*)d_ws;
    _Float16* wh = xh + nX;
    _Float16* qh = wh + 3 * nW;
    _Float16* kh = qh + nX;
    _Float16* vT = kh + nX;

    // one-pass fp32 -> f16 conversions (removes all redundant cvt work)
    {
        const int n8x = (int)(nX / 8);
        const int n8w = (int)(nW / 8);
        cvt_f16_kernel<<<(n8x + 255) / 256, 256, 0, stream>>>(x,  xh, n8x);
        cvt_f16_kernel<<<(n8w + 255) / 256, 256, 0, stream>>>(Wq, wh,          n8w);
        cvt_f16_kernel<<<(n8w + 255) / 256, 256, 0, stream>>>(Wk, wh + nW,     n8w);
        cvt_f16_kernel<<<(n8w + 255) / 256, 256, 0, stream>>>(Wv, wh + 2 * nW, n8w);
    }

    dim3 gproj(6, (BATCH * TDIM) / 16, 3);
    qkv_proj_kernel<<<gproj, 256, 0, stream>>>(xh, wh, qh, kh, vT);

    dim3 gattn(TDIM / 16, BATCH);
    flash_attn_kernel<<<gattn, 32, 0, stream>>>(qh, kh, vT, out);
}
